// RelationNet_13615046328919
// MI455X (gfx1250) — matrix operand order repaired and checked by reading, not compiled, not run on hardware
//
#include <hip/hip_runtime.h>
#include <hip/hip_bf16.h>

// ---------------------------------------------------------------------------
// RelationNet forward for MI455X (gfx1250, wave32, WMMA bf16 16x16x32).
// Dominant cost: three (204800 x 256) x (256 x 256) GEMMs -> v_wmma path.
// Round 3: double-buffered async A staging (copy/compute overlap via
// ASYNCcnt), batched B-fragment ds loads per K-step, hoisted bias.
// ---------------------------------------------------------------------------

#define RN_ASYNC 1   // use global_load_async_to_lds_b128 for A-tile staging

typedef __attribute__((ext_vector_type(16))) __bf16 bf16x16;
typedef __attribute__((ext_vector_type(8)))  __bf16 bf16x8;
typedef __attribute__((ext_vector_type(8)))  float  f32x8;

#define B_   512
#define S_   20
#define MSL_ 32
#define MQL_ 16
#define E_   32
#define H_   32
#define G_   256
#define A_   1000
#define NPAIR (S_ * S_)          // 400
#define MTOT  (B_ * NPAIR)       // 204800 rows in relation GEMMs

#define LDW_  264                // padded row stride (bf16 elems): 528B rows,
                                 // row*132 words -> 4-bank stride, no conflicts
#define GEMM_THREADS 512         // 16 waves
#define GEMM_MTILE   128         // rows per workgroup iteration
#define GEMM_GRID    512         // persistent workgroups
#define GEMM_NTILES  (MTOT / GEMM_MTILE)   // 1600

// ---------------------------------------------------------------------------
// WMMA fragment loaders (wave32).
// ---------------------------------------------------------------------------
union ABFrag { bf16x16 v; bf16x8 h[2]; };

// A (16x32 bf16, M x K): lane<16 -> row m=lane,    K = {0..7, 16..23}
//                        lane>=16 -> row m=lane-16, K = {8..15, 24..31}
__device__ inline bf16x16 load_a_frag_lds(const __bf16* base, int ld, int lane) {
    int m  = lane & 15;
    int kb = (lane >> 4) << 3;           // 0 or 8
    const __bf16* p = base + m * ld + kb;
    ABFrag f;
    f.h[0] = *(const bf16x8*)(p);        // K = kb .. kb+7
    f.h[1] = *(const bf16x8*)(p + 16);   // K = kb+16 .. kb+23
    return f.v;
}

// B (32x16 bf16, K x N) taken from row-major W[nOut][kIn]:
// lane<16 -> col n=lane, K={0..7,16..23} ; lane>=16 -> col n=lane-16, K={8..15,24..31}.
__device__ inline bf16x16 load_b_frag(const __bf16* __restrict__ Wp, int ldw,
                                      int n0, int k0, int lane) {
    int n  = lane & 15;
    int kb = (lane >> 4) << 3;           // 0 or 8
    const __bf16* p = Wp + (long)(n0 + n) * ldw + k0 + kb;
    ABFrag f;
    f.h[0] = *(const bf16x8*)(p);
    f.h[1] = *(const bf16x8*)(p + 16);
    return f.v;
}

// Stage one 128x256 bf16 A tile into a padded LDS buffer (async DMA path).
__device__ inline void stage_a_tile(__bf16* dst, const __bf16* __restrict__ src,
                                    int tid) {
#if RN_ASYNC
    for (int idx = tid; idx < GEMM_MTILE * 32; idx += GEMM_THREADS) {
        int r = idx >> 5, s = idx & 31;
        unsigned loff = (unsigned)(size_t)(dst + r * LDW_ + s * 8);
        unsigned long gaddr =
            (unsigned long)(const void*)(src + r * G_ + s * 8);
        asm volatile("global_load_async_to_lds_b128 %0, %1, off"
                     :: "v"(loff), "v"(gaddr) : "memory");
    }
#else
    for (int idx = tid; idx < GEMM_MTILE * 32; idx += GEMM_THREADS) {
        int r = idx >> 5, s = idx & 31;
        *(bf16x8*)(dst + r * LDW_ + s * 8) =
            *(const bf16x8*)(src + r * G_ + s * 8);
    }
#endif
}

__device__ inline void wait_async_copies() {
#if RN_ASYNC
    asm volatile("s_wait_asynccnt 0" ::: "memory");
#endif
}

// ---------------------------------------------------------------------------
// Relation-core GEMM:  Xout = relu(Xin @ W^T + bias), bf16 in/out, f32 acc.
// Persistent grid-stride over M-tiles; full weight matrix resident in LDS;
// A tiles double-buffered with async DMA prefetch overlapping compute.
// Block = 512 threads = 16 waves; wave tile 16(M) x 128(N).
// ---------------------------------------------------------------------------
__global__ __launch_bounds__(GEMM_THREADS) void rn_gemm_kernel(
    const __bf16* __restrict__ Xin, const __bf16* __restrict__ W,
    const float* __restrict__ bias, __bf16* __restrict__ Xout, int ntiles)
{
    __shared__ __align__(16) __bf16 Wlds[G_][LDW_];             // 132 KB
    __shared__ __align__(16) __bf16 Atile[2][GEMM_MTILE][LDW_]; // 2 x 66 KB

    const int tid = threadIdx.x;

    // Stage full 256x256 bf16 weight matrix into padded LDS (once).
    for (int idx = tid; idx < G_ * 32; idx += GEMM_THREADS) {
        int r = idx >> 5, s = idx & 31;
        *(bf16x8*)&Wlds[r][s * 8] = *(const bf16x8*)(W + r * G_ + s * 8);
    }

    const int wave = tid >> 5, lane = tid & 31;
    const int m16  = wave >> 1;            // 0..7
    const int n0   = (wave & 1) * 128;     // 0 or 128
    const int n_lane = lane & 15;
    const int mh     = (lane >> 4) * 8;

    // Loop-invariant bias for this wave's 8 N-tiles.
    float bv[8];
#pragma unroll
    for (int nt = 0; nt < 8; ++nt) bv[nt] = bias[n0 + nt * 16 + n_lane];

    // Prologue: issue async staging of the first tile into buffer 0.
    if ((int)blockIdx.x < ntiles)
        stage_a_tile(&Atile[0][0][0],
                     Xin + (long)blockIdx.x * GEMM_MTILE * G_, tid);

    int cur = 0;
    for (int tile = blockIdx.x; tile < ntiles; tile += gridDim.x) {
        // Current buffer's DMA done (per-wave), then block-wide sync.
        wait_async_copies();
        __syncthreads();

        // Prefetch next tile into the other buffer (its last readers all
        // passed the barrier above, so the DMA cannot race them).
        int pf = tile + gridDim.x;
        if (pf < ntiles)
            stage_a_tile(&Atile[cur ^ 1][0][0],
                         Xin + (long)pf * GEMM_MTILE * G_, tid);

        const long mbase = (long)tile * GEMM_MTILE;

        // ---- 16x128 output per wave: 8 K-steps x 8 N-tiles of WMMA ----
        f32x8 acc[8];
#pragma unroll
        for (int nt = 0; nt < 8; ++nt) acc[nt] = {};

#pragma unroll
        for (int kk = 0; kk < 8; ++kk) {   // K = 256 in steps of 32
            const __bf16* abase = &Atile[cur][m16 * 16][kk * 32];
            bf16x16 a = load_a_frag_lds(abase, LDW_, lane);
            // Batch all 8 B fragments (one ds clause, one wait)...
            bf16x16 bf[8];
#pragma unroll
            for (int nt = 0; nt < 8; ++nt)
                bf[nt] = load_b_frag(&Wlds[0][0], LDW_, n0 + nt * 16,
                                     kk * 32, lane);
            // ...then run the WMMAs back-to-back.
#pragma unroll
            for (int nt = 0; nt < 8; ++nt)
                acc[nt] = __builtin_amdgcn_wmma_f32_16x16x32_bf16(
                    false, a, false, bf[nt], (short)0, acc[nt], false, false);
        }

        // ---- Epilogue: bias + ReLU + bf16 store ----
        // C/D layout: lane&15 = N, VGPR r -> M = r + 8*(lane>>4).
#pragma unroll
        for (int nt = 0; nt < 8; ++nt) {
            int n = n0 + nt * 16 + n_lane;
#pragma unroll
            for (int r = 0; r < 8; ++r) {
                float v = acc[nt][r] + bv[nt];
                v = v > 0.f ? v : 0.f;
                Xout[(mbase + m16 * 16 + mh + r) * G_ + n] = (__bf16)v;
            }
        }
        cur ^= 1;
    }
}

// ---------------------------------------------------------------------------
// LSTM: wave per sequence, lane per hidden unit (H = 32 = wave32).
// Gates order per reference split: [i | f | g | o].
// ---------------------------------------------------------------------------
__global__ __launch_bounds__(256) void lstm_kernel(
    const int* __restrict__ tokens, int T,
    const float* __restrict__ emb,
    const float* __restrict__ Wih, const float* __restrict__ Whh,
    const float* __restrict__ bih, const float* __restrict__ bhh,
    float* __restrict__ hout, int nseq)
{
    __shared__ float sWih[128][33];   // padded to kill bank conflicts
    __shared__ float sWhh[128][33];
    __shared__ float sb[128];
    __shared__ float xs[8][32];
    __shared__ float hs[8][32];

    const int tid = threadIdx.x;
    for (int idx = tid; idx < 128 * 32; idx += 256) {
        int r = idx >> 5, c2 = idx & 31;
        sWih[r][c2] = Wih[idx];
        sWhh[r][c2] = Whh[idx];
    }
    if (tid < 128) sb[tid] = bih[tid] + bhh[tid];

    const int wave = tid >> 5, lane = tid & 31;
    const int seq  = blockIdx.x * 8 + wave;
    const bool active = seq < nseq;

    float c = 0.f;
    hs[wave][lane] = 0.f;
    __syncthreads();

    for (int t = 0; t < T; ++t) {
        int tok = active ? tokens[seq * T + t] : 0;
        xs[wave][lane] = emb[tok * E_ + lane];
        __syncthreads();

        float a0 = sb[lane], a1 = sb[32 + lane];
        float a2 = sb[64 + lane], a3 = sb[96 + lane];
#pragma unroll 8
        for (int e = 0; e < 32; ++e) {
            float xv = xs[wave][e];
            float hv = hs[wave][e];
            a0 += sWih[lane][e]      * xv + sWhh[lane][e]      * hv;
            a1 += sWih[32 + lane][e] * xv + sWhh[32 + lane][e] * hv;
            a2 += sWih[64 + lane][e] * xv + sWhh[64 + lane][e] * hv;
            a3 += sWih[96 + lane][e] * xv + sWhh[96 + lane][e] * hv;
        }
        __syncthreads();

        float ig = 1.f / (1.f + __expf(-a0));
        float fg = 1.f / (1.f + __expf(-a1));
        float gg = tanhf(a2);
        float og = 1.f / (1.f + __expf(-a3));
        c = fg * c + ig * gg;
        hs[wave][lane] = og * tanhf(c);
    }
    __syncthreads();
    if (active) hout[seq * H_ + lane] = hs[wave][lane];
}

// ---------------------------------------------------------------------------
// g1 projections (obj = [h(32) | onehot_i(20)] -> tag part is a column pick).
// ---------------------------------------------------------------------------
__global__ __launch_bounds__(256) void pipj_kernel(
    const float* __restrict__ s_state, const float* __restrict__ g1w,
    float* __restrict__ pi, float* __restrict__ pj)
{
    __shared__ float sv[32];
    const int blk = blockIdx.x;        // b*20 + i
    const int i   = blk % S_;
    const int g   = threadIdx.x;
    if (g < 32) sv[g] = s_state[blk * H_ + g];
    __syncthreads();

    const float* wrow = g1w + g * 136;
    float accA = wrow[32 + i];
    float accB = wrow[52 + 32 + i];
#pragma unroll 8
    for (int h = 0; h < 32; ++h) {
        float s = sv[h];
        accA += s * wrow[h];
        accB += s * wrow[52 + h];
    }
    pi[blk * G_ + g] = accA;
    pj[blk * G_ + g] = accB;
}

__global__ __launch_bounds__(256) void pq_kernel(
    const float* __restrict__ q_state, const float* __restrict__ g1w,
    const float* __restrict__ g1b, float* __restrict__ pqb)
{
    __shared__ float qv[32];
    const int b = blockIdx.x, g = threadIdx.x;
    if (g < 32) qv[g] = q_state[b * H_ + g];
    __syncthreads();
    const float* wrow = g1w + g * 136 + 104;
    float acc = g1b[g];
#pragma unroll 8
    for (int h = 0; h < 32; ++h) acc += qv[h] * wrow[h];
    pqb[b * G_ + g] = acc;
}

// X0[p,g] = bf16(relu(pi[b,i,g] + pj[b,j,g] + pqb[b,g])), p = b*400+i*20+j
__global__ __launch_bounds__(256) void x0_kernel(
    const float* __restrict__ pi, const float* __restrict__ pj,
    const float* __restrict__ pqb, __bf16* __restrict__ x0)
{
    long idx = (long)blockIdx.x * 256 + threadIdx.x;   // < 52,428,800
    int  g = (int)(idx & 255);
    long p = idx >> 8;
    int  j = (int)(p % S_);
    long q = p / S_;        // b*20 + i
    long b = q / S_;
    float v = pi[q * G_ + g] + pj[(b * S_ + j) * G_ + g] + pqb[b * G_ + g];
    x0[idx] = (__bf16)(v > 0.f ? v : 0.f);
}

__global__ __launch_bounds__(256) void cvt_bf16_kernel(
    const float* __restrict__ src, __bf16* __restrict__ dst, int n)
{
    int i = blockIdx.x * 256 + threadIdx.x;
    if (i < n) dst[i] = (__bf16)src[i];
}

// sumg[b,g] = sum_{p<400} x4[b*400+p, g]
__global__ __launch_bounds__(256) void pairsum_kernel(
    const __bf16* __restrict__ x4, float* __restrict__ sumg)
{
    const int b = blockIdx.x, g = threadIdx.x;
    const __bf16* p = x4 + (long)b * NPAIR * G_ + g;
    float acc = 0.f;
    for (int t = 0; t < NPAIR; ++t) acc += (float)p[(long)t * G_];
    sumg[b * G_ + g] = acc;
}

// Dense FC: out[r,o] = act(sum_k in[r,k]*W[o,k] + b[o]); grid (rows, O-tiles)
__global__ __launch_bounds__(256) void fc_kernel(
    const float* __restrict__ in, const float* __restrict__ W,
    const float* __restrict__ bias, float* __restrict__ out,
    int K, int O, int do_relu)
{
    __shared__ float row[512];
    const int r = blockIdx.x, tid = threadIdx.x;
    for (int k = tid; k < K; k += 256) row[k] = in[(long)r * K + k];
    __syncthreads();
    int o = blockIdx.y * 256 + tid;
    if (o < O) {
        const float* wrow = W + (long)o * K;
        float acc = bias[o];
        for (int k = 0; k < K; ++k) acc += row[k] * wrow[k];
        if (do_relu) acc = acc > 0.f ? acc : 0.f;
        out[(long)r * O + o] = acc;
    }
}

// ---------------------------------------------------------------------------
// Host launcher
// ---------------------------------------------------------------------------
extern "C" void kernel_launch(void* const* d_in, const int* in_sizes, int n_in,
                              void* d_out, int out_size, void* d_ws, size_t ws_size,
                              hipStream_t stream) {
    (void)in_sizes; (void)n_in; (void)out_size; (void)ws_size;

    const int*   story    = (const int*)  d_in[0];
    const int*   question = (const int*)  d_in[1];
    const float* emb      = (const float*)d_in[2];
    const float* s_Wih    = (const float*)d_in[3];
    const float* s_Whh    = (const float*)d_in[4];
    const float* s_bih    = (const float*)d_in[5];
    const float* s_bhh    = (const float*)d_in[6];
    const float* q_Wih    = (const float*)d_in[7];
    const float* q_Whh    = (const float*)d_in[8];
    const float* q_bih    = (const float*)d_in[9];
    const float* q_bhh    = (const float*)d_in[10];
    const float* g1_w     = (const float*)d_in[11];
    const float* g1_b     = (const float*)d_in[12];
    const float* g2_w     = (const float*)d_in[13];
    const float* g2_b     = (const float*)d_in[14];
    const float* g3_w     = (const float*)d_in[15];
    const float* g3_b     = (const float*)d_in[16];
    const float* g4_w     = (const float*)d_in[17];
    const float* g4_b     = (const float*)d_in[18];
    const float* f1_w     = (const float*)d_in[19];
    const float* f1_b     = (const float*)d_in[20];
    const float* f2_w     = (const float*)d_in[21];
    const float* f2_b     = (const float*)d_in[22];
    const float* f3_w     = (const float*)d_in[23];
    const float* f3_b     = (const float*)d_in[24];
    float* out = (float*)d_out;

    // Deterministic workspace carve-up (256B aligned).
    char* ws = (char*)d_ws;
    size_t off = 0;
    auto take = [&](size_t bytes) -> char* {
        char* p = ws + off;
        off = (off + bytes + 255) & ~(size_t)255;
        return p;
    };
    float*  s_state = (float*) take((size_t)B_ * S_ * H_ * 4);      // 10240x32
    float*  q_state = (float*) take((size_t)B_ * H_ * 4);           // 512x32
    float*  pi      = (float*) take((size_t)B_ * S_ * G_ * 4);      // 10240x256
    float*  pj      = (float*) take((size_t)B_ * S_ * G_ * 4);
    float*  pqb     = (float*) take((size_t)B_ * G_ * 4);
    __bf16* w2b     = (__bf16*)take((size_t)G_ * G_ * 2);
    __bf16* w3b     = (__bf16*)take((size_t)G_ * G_ * 2);
    __bf16* w4b     = (__bf16*)take((size_t)G_ * G_ * 2);
    __bf16* Xa      = (__bf16*)take((size_t)MTOT * G_ * 2);         // 100 MB
    __bf16* Xb      = (__bf16*)take((size_t)MTOT * G_ * 2);         // 100 MB
    float*  sumg    = (float*) take((size_t)B_ * G_ * 4);
    float*  h1      = (float*) take((size_t)B_ * G_ * 4);
    float*  h2      = (float*) take((size_t)B_ * 512 * 4);

    // 1) LSTMs (story: 10240 seqs T=32; question: 512 seqs T=16)
    lstm_kernel<<<(B_ * S_) / 8, 256, 0, stream>>>(
        story, MSL_, emb, s_Wih, s_Whh, s_bih, s_bhh, s_state, B_ * S_);
    lstm_kernel<<<B_ / 8, 256, 0, stream>>>(
        question, MQL_, emb, q_Wih, q_Whh, q_bih, q_bhh, q_state, B_);

    // 2) g1 projections
    pipj_kernel<<<B_ * S_, 256, 0, stream>>>(s_state, g1_w, pi, pj);
    pq_kernel<<<B_, 256, 0, stream>>>(q_state, g1_w, g1_b, pqb);

    // 3) bf16 weight copies for the WMMA GEMMs
    cvt_bf16_kernel<<<(G_ * G_) / 256, 256, 0, stream>>>(g2_w, w2b, G_ * G_);
    cvt_bf16_kernel<<<(G_ * G_) / 256, 256, 0, stream>>>(g3_w, w3b, G_ * G_);
    cvt_bf16_kernel<<<(G_ * G_) / 256, 256, 0, stream>>>(g4_w, w4b, G_ * G_);

    // 4) X0 = relu(pi_i + pj_j + pq + b1) in bf16
    x0_kernel<<<(MTOT * G_) / 256, 256, 0, stream>>>(pi, pj, pqb, Xa);

    // 5) The three WMMA relation GEMMs (the 80-GFLOP core), persistent blocks
    rn_gemm_kernel<<<GEMM_GRID, GEMM_THREADS, 0, stream>>>(
        Xa, w2b, g2_b, Xb, GEMM_NTILES);
    rn_gemm_kernel<<<GEMM_GRID, GEMM_THREADS, 0, stream>>>(
        Xb, w3b, g3_b, Xa, GEMM_NTILES);
    rn_gemm_kernel<<<GEMM_GRID, GEMM_THREADS, 0, stream>>>(
        Xa, w4b, g4_b, Xb, GEMM_NTILES);

    // 6) Sum over pairs, then the FC head
    pairsum_kernel<<<B_, 256, 0, stream>>>(Xb, sumg);
    fc_kernel<<<dim3(B_, 1), 256, 0, stream>>>(sumg, f1_w, f1_b, h1, G_, G_, 1);
    fc_kernel<<<dim3(B_, 2), 256, 0, stream>>>(h1, f2_w, f2_b, h2, G_, 512, 1);
    fc_kernel<<<dim3(B_, 4), 256, 0, stream>>>(h2, f3_w, f3_b, out, 512, A_, 0);
}
